// Block_3530463117419
// MI455X (gfx1250) — matrix operand order repaired and checked by reading, not compiled, not run on hardware
//
#include <hip/hip_runtime.h>

#define B_  2
#define T_  2048
#define C_  768
#define H_  12
#define HD_ 64
#define D2_ 32
#define FF_ 3072
#define BT_ (B_ * T_)

typedef __attribute__((ext_vector_type(16))) __bf16 v16bf;
typedef __attribute__((ext_vector_type(8)))  float  v8f;
typedef __attribute__((ext_vector_type(4)))  unsigned int v4u;
typedef __attribute__((ext_vector_type(4)))  int v4i;
typedef __attribute__((ext_vector_type(8)))  int v8i;

#define USE_TDM (__has_builtin(__builtin_amdgcn_tensor_load_to_lds) && \
                 __has_builtin(__builtin_amdgcn_s_wait_tensorcnt))

union FragU {
    v16bf v;
    uint4 q[2];
    unsigned short s[16];
};

static __device__ inline unsigned short f32_to_bf16(float f) {
    union { float f; unsigned u; } x;
    x.f = f;
    unsigned r = x.u + 0x7FFFu + ((x.u >> 16) & 1u);  // round to nearest even
    return (unsigned short)(r >> 16);
}

// A-fragment (16x32 bf16): lane m=L&15 holds row m; half h=L>>4.
// element e -> K = h*16 + e  => one contiguous 16-elem chunk.
static __device__ inline v16bf load_a_frag(const unsigned short* base, int stride, int lane) {
    FragU f;
    const int m = lane & 15, h = lane >> 4;
    const unsigned short* p = base + (size_t)m * stride + h * 16;
    f.q[0] = *reinterpret_cast<const uint4*>(p);
    f.q[1] = *reinterpret_cast<const uint4*>(p + 8);
    return f.v;
}

// B-fragment (32x16 bf16) from row-major W^T (N x K):
// lane n=L&15 holds column n; half h=L>>4; element e -> K = h*16 + e.
static __device__ inline v16bf load_b_frag(const unsigned short* base, int stride, int lane) {
    FragU f;
    const int n = lane & 15, h = lane >> 4;
    const unsigned short* p = base + (size_t)n * stride + h * 16;
    f.q[0] = *reinterpret_cast<const uint4*>(p);
    f.q[1] = *reinterpret_cast<const uint4*>(p + 8);
    return f.v;
}

#if USE_TDM
// 2D TDM descriptor load: tile_rows x tile_k (bf16) from row-major tensor
// with row stride row_stride_elems, into LDS at lds_byte_addr.
// D# layout per CDNA5 ISA ch.8 (group0 128b, group1 256b).
static __device__ inline void tdm_load_tile_2d(const unsigned short* gsrc,
                                               unsigned lds_byte_addr,
                                               int tile_k, int tile_rows,
                                               int row_stride_elems) {
    unsigned long long ga = (unsigned long long)(const void*)gsrc;
    v4u g0;
    g0[0] = 1u;                                    // count=1, user descriptor
    g0[1] = lds_byte_addr;                         // lds_addr
    g0[2] = (unsigned)ga;                          // global_addr[31:0]
    g0[3] = (unsigned)((ga >> 32) & 0x01FFFFFFu)   // global_addr[56:32]
          | (2u << 30);                            // type=2 ("image")
    const unsigned td0 = (unsigned)row_stride_elems;  // tensor_dim0 >= tile_k
    const unsigned td1 = 0x7FFFu;                     // tensor_dim1 (rows), large
    v8i g1;
    g1[0] = (int)(1u << 16);                          // data_size=1 => 2 bytes
    g1[1] = (int)((td0 & 0xFFFFu) << 16);             // tensor_dim0[15:0]
    g1[2] = (int)(((td0 >> 16) & 0xFFFFu) | ((td1 & 0xFFFFu) << 16));
    g1[3] = (int)(((td1 >> 16) & 0xFFFFu) | (((unsigned)tile_k & 0xFFFFu) << 16));
    g1[4] = (int)((unsigned)tile_rows & 0xFFFFu);     // tile_dim1; tile_dim2=0
    g1[5] = (int)td0;                                 // tensor_dim0_stride[31:0]
    g1[6] = 0;                                        // stride[47:32]; dim1_stride lo
    g1[7] = 0;
    v4i z4 = {0, 0, 0, 0};
#if __clang_major__ >= 23
    v8i z8 = {0, 0, 0, 0, 0, 0, 0, 0};
    __builtin_amdgcn_tensor_load_to_lds(g0, g1, z4, z4, z8, 0);
#else
    __builtin_amdgcn_tensor_load_to_lds(g0, g1, z4, z4, 0);
#endif
}
#endif

// ---------------------------------------------------------------------------
// dst[n*K + k] = bf16(src[k*N + n])   (weight transpose + downconvert)
// ---------------------------------------------------------------------------
__global__ void transpose_bf16(const float* __restrict__ src,
                               unsigned short* __restrict__ dst, int K, int N) {
    int idx = blockIdx.x * 256 + threadIdx.x;
    if (idx >= K * N) return;
    int n = idx / K;
    int k = idx - n * K;
    dst[idx] = f32_to_bf16(src[(size_t)k * N + n]);
}

// ---------------------------------------------------------------------------
// RMSNorm over C=768, one 256-thread block per row, bf16 output
// ---------------------------------------------------------------------------
__global__ void rmsnorm_bf16(const float* __restrict__ x, const float* __restrict__ g,
                             unsigned short* __restrict__ out) {
    __shared__ float red[8];
    const int row = blockIdx.x;
    const float* xr = x + (size_t)row * C_;
    const int t = threadIdx.x;
    float v0 = xr[t], v1 = xr[t + 256], v2 = xr[t + 512];
    float ss = v0 * v0 + v1 * v1 + v2 * v2;
#pragma unroll
    for (int off = 16; off > 0; off >>= 1) ss += __shfl_xor(ss, off);
    if ((t & 31) == 0) red[t >> 5] = ss;
    __syncthreads();
    float tot = 0.f;
#pragma unroll
    for (int i = 0; i < 8; ++i) tot += red[i];
    float rinv = rsqrtf(tot * (1.0f / (float)C_) + 1e-5f);
    unsigned short* o = out + (size_t)row * C_;
    o[t]       = f32_to_bf16(v0 * rinv * g[t]);
    o[t + 256] = f32_to_bf16(v1 * rinv * g[t + 256]);
    o[t + 512] = f32_to_bf16(v2 * rinv * g[t + 512]);
}

// ---------------------------------------------------------------------------
// WMMA GEMM: Out[M,N] = A_bf16[M,K] @ Bt_bf16[N,K]^T
// Block = 8 waves = 128 rows x 64 cols. Each wave: one 16x64 strip (4 acc).
// B tile (64x32 bf16, 4KB) staged into LDS via TDM (double-buffered),
// shared by all 8 waves; A fragments stream from global.
// mode 0: f32 store   mode 1: f32 + residual R   mode 2: bf16(relu(v)^2)
// ---------------------------------------------------------------------------
__global__ void wmma_gemm(const unsigned short* __restrict__ A,
                          const unsigned short* __restrict__ Bt,
                          const float* __restrict__ R, void* __restrict__ Out,
                          int M, int N, int K, int mode) {
    __shared__ unsigned short bsh[2][64 * 32];   // double-buffered B tile
    const int lane = threadIdx.x & 31;
    const int wave = threadIdx.x >> 5;
    const int nbase = blockIdx.x * 64;
    const int mbase = blockIdx.y * 128 + wave * 16;

    v8f acc[4];
#pragma unroll
    for (int t = 0; t < 4; ++t) acc[t] = (v8f){0.f,0.f,0.f,0.f,0.f,0.f,0.f,0.f};

    const unsigned short* Arow = A + (size_t)mbase * K;
    const unsigned short* Btile = Bt + (size_t)nbase * K;
    const int S = K / 32;

    // stage K-step 0 into buffer 0
#if USE_TDM
    if (wave == 0)
        tdm_load_tile_2d(Btile, (unsigned)(size_t)&bsh[0][0], 32, 64, K);
#else
    {
        int row = threadIdx.x >> 2, c = (threadIdx.x & 3) * 8;
        *(uint4*)&bsh[0][row * 32 + c] = *(const uint4*)&Btile[(size_t)row * K + c];
    }
#endif

    for (int s = 0; s < S; ++s) {
        const int kb = s * 32;
#if USE_TDM
        if (wave == 0) __builtin_amdgcn_s_wait_tensorcnt(0);   // buf[s&1] ready
#endif
        __syncthreads();
        if (s + 1 < S) {                                        // prefetch next
#if USE_TDM
            if (wave == 0)
                tdm_load_tile_2d(Btile + kb + 32,
                                 (unsigned)(size_t)&bsh[(s + 1) & 1][0], 32, 64, K);
#else
            int row = threadIdx.x >> 2, c = (threadIdx.x & 3) * 8;
            *(uint4*)&bsh[(s + 1) & 1][row * 32 + c] =
                *(const uint4*)&Btile[(size_t)row * K + kb + 32 + c];
#endif
        }
        __builtin_prefetch(Arow + kb + 256, 0, 3);
        v16bf a = load_a_frag(Arow + kb, K, lane);              // shared A frag
        const unsigned short* bb = &bsh[s & 1][0];
#pragma unroll
        for (int t = 0; t < 4; ++t) {
            v16bf b = load_b_frag(bb + t * 16 * 32, 32, lane);  // ds_load_b128
            acc[t] = __builtin_amdgcn_wmma_f32_16x16x32_bf16(false, a, false, b,
                                                             (short)0, acc[t],
                                                             false, false);
        }
    }

    const int hh = lane >> 4;
    const int n = lane & 15;
#pragma unroll
    for (int t = 0; t < 4; ++t) {
#pragma unroll
        for (int r = 0; r < 8; ++r) {
            int row = mbase + r + hh * 8;
            size_t idx = (size_t)row * N + nbase + t * 16 + n;
            float v = acc[t][r];
            if (mode == 0) {
                ((float*)Out)[idx] = v;
            } else if (mode == 1) {
                ((float*)Out)[idx] = v + R[idx];
            } else {
                float u = v > 0.f ? v : 0.f;
                ((unsigned short*)Out)[idx] = f32_to_bf16(u * u);
            }
        }
    }
}

// ---------------------------------------------------------------------------
// RoPE on q,k (f32 in, bf16 out); 1/sqrt(HD) folded into q
// ---------------------------------------------------------------------------
__global__ void rope_qk(const float* __restrict__ qf, const float* __restrict__ kf,
                        const float* __restrict__ cosb, const float* __restrict__ sinb,
                        unsigned short* __restrict__ qb, unsigned short* __restrict__ kb) {
    int idx = blockIdx.x * 256 + threadIdx.x;   // B*T*H*D2 threads
    int d = idx & 31;
    int bth = idx >> 5;                         // (b*T + t)*H + h
    int t = (bth / H_) % T_;
    size_t base = (size_t)bth * HD_;
    float c = cosb[t * D2_ + d], s = sinb[t * D2_ + d];
    float q1 = qf[base + d], q2 = qf[base + d + D2_];
    float k1 = kf[base + d], k2 = kf[base + d + D2_];
    const float scale = 0.125f;                 // 1/sqrt(64)
    qb[base + d]       = f32_to_bf16((q1 * c + q2 * s) * scale);
    qb[base + d + D2_] = f32_to_bf16((-q1 * s + q2 * c) * scale);
    kb[base + d]       = f32_to_bf16(k1 * c + k2 * s);
    kb[base + d + D2_] = f32_to_bf16(-k1 * s + k2 * c);
}

__global__ void cvt_bf16(const float* __restrict__ src, unsigned short* __restrict__ dst, int n) {
    int i = blockIdx.x * 256 + threadIdx.x;
    if (i < n) dst[i] = f32_to_bf16(src[i]);
}

// ---------------------------------------------------------------------------
// Flash attention: grid (B*H, T/128), 8 waves/block, 16 q-rows per wave.
// Per 32-key step: 4 WMMAs for scores, online softmax with 16-lane shuffles,
// P staged via LDS into A-frag layout, 4 WMMAs for P·V.
// ---------------------------------------------------------------------------
__global__ void attn_kernel(const unsigned short* __restrict__ qb,
                            const unsigned short* __restrict__ kb,
                            const unsigned short* __restrict__ vb,
                            unsigned short* __restrict__ yb) {
    __shared__ unsigned short pl[8][16 * 32];   // per-wave P tile (16x32 bf16)

    const int lane = threadIdx.x & 31;
    const int wave = threadIdx.x >> 5;
    const int bh   = blockIdx.x;
    const int b    = bh / H_, head = bh % H_;
    const int qbase = blockIdx.y * 128 + wave * 16;
    const int nn = lane & 15, hh = lane >> 4;
    const int stride = H_ * HD_;                // row stride in (B,T,H,HD)

    const unsigned short* qrow = qb + (((size_t)b * T_ + qbase) * H_ + head) * HD_;
    v16bf qa0 = load_a_frag(qrow, stride, lane);
    v16bf qa1 = load_a_frag(qrow + 32, stride, lane);

    v8f acc0 = {0.f,0.f,0.f,0.f,0.f,0.f,0.f,0.f};
    v8f acc1 = acc0, acc2 = acc0, acc3 = acc0;
    float mrow[8], lrow[8];
#pragma unroll
    for (int r = 0; r < 8; ++r) { mrow[r] = -INFINITY; lrow[r] = 0.f; }

    const int jend = qbase + 16;                // causal bound (exclusive)
    for (int j = 0; j < jend; j += 32) {
        const unsigned short* krow0 = kb + (((size_t)b * T_ + j) * H_ + head) * HD_;
        const unsigned short* krow1 = krow0 + (size_t)16 * stride;

        v8f s0 = {0.f,0.f,0.f,0.f,0.f,0.f,0.f,0.f};
        v8f s1 = s0;
        s0 = __builtin_amdgcn_wmma_f32_16x16x32_bf16(false, qa0, false,
                 load_b_frag(krow0, stride, lane), (short)0, s0, false, false);
        s0 = __builtin_amdgcn_wmma_f32_16x16x32_bf16(false, qa1, false,
                 load_b_frag(krow0 + 32, stride, lane), (short)0, s0, false, false);
        s1 = __builtin_amdgcn_wmma_f32_16x16x32_bf16(false, qa0, false,
                 load_b_frag(krow1, stride, lane), (short)0, s1, false, false);
        s1 = __builtin_amdgcn_wmma_f32_16x16x32_bf16(false, qa1, false,
                 load_b_frag(krow1 + 32, stride, lane), (short)0, s1, false, false);

        float p0[8], p1[8];
#pragma unroll
        for (int r = 0; r < 8; ++r) {
            int qg = qbase + r + hh * 8;
            int k0 = j + nn, k1 = k0 + 16;
            float a = (k0 <= qg) ? s0[r] : -INFINITY;
            float bbv = (k1 <= qg) ? s1[r] : -INFINITY;
            float mx = fmaxf(a, bbv);
#pragma unroll
            for (int off = 1; off < 16; off <<= 1) mx = fmaxf(mx, __shfl_xor(mx, off));
            float mnew = fmaxf(mrow[r], mx);
            float alpha = __expf(mrow[r] - mnew);
            float e0 = __expf(a - mnew);
            float e1 = __expf(bbv - mnew);
            float sum = e0 + e1;
#pragma unroll
            for (int off = 1; off < 16; off <<= 1) sum += __shfl_xor(sum, off);
            lrow[r] = lrow[r] * alpha + sum;
            mrow[r] = mnew;
            p0[r] = e0; p1[r] = e1;
            acc0[r] *= alpha; acc1[r] *= alpha; acc2[r] *= alpha; acc3[r] *= alpha;
        }

        // P (C/D layout) -> LDS row-major 16x32 -> A fragment
#pragma unroll
        for (int r = 0; r < 8; ++r) {
            int row = r + hh * 8;
            pl[wave][row * 32 + nn]      = f32_to_bf16(p0[r]);
            pl[wave][row * 32 + 16 + nn] = f32_to_bf16(p1[r]);
        }
        v16bf pf = load_a_frag(&pl[wave][0], 32, lane);

        // V B-fragments (K = key dim, strided gather), acc += P @ V
#pragma unroll
        for (int t = 0; t < 4; ++t) {
            FragU vf;
            int nhd = t * 16 + nn;
            const unsigned short* vbase =
                vb + (((size_t)b * T_ + j + hh * 16) * H_ + head) * HD_ + nhd;
#pragma unroll
            for (int e = 0; e < 16; ++e) vf.s[e] = vbase[(size_t)e * stride];
            if (t == 0) acc0 = __builtin_amdgcn_wmma_f32_16x16x32_bf16(false, pf, false, vf.v, (short)0, acc0, false, false);
            if (t == 1) acc1 = __builtin_amdgcn_wmma_f32_16x16x32_bf16(false, pf, false, vf.v, (short)0, acc1, false, false);
            if (t == 2) acc2 = __builtin_amdgcn_wmma_f32_16x16x32_bf16(false, pf, false, vf.v, (short)0, acc2, false, false);
            if (t == 3) acc3 = __builtin_amdgcn_wmma_f32_16x16x32_bf16(false, pf, false, vf.v, (short)0, acc3, false, false);
        }
    }

#pragma unroll
    for (int r = 0; r < 8; ++r) {
        int row = qbase + r + hh * 8;
        float inv = 1.0f / lrow[r];
        size_t ob = (size_t)(b * T_ + row) * C_ + head * HD_ + nn;
        yb[ob +  0] = f32_to_bf16(acc0[r] * inv);
        yb[ob + 16] = f32_to_bf16(acc1[r] * inv);
        yb[ob + 32] = f32_to_bf16(acc2[r] * inv);
        yb[ob + 48] = f32_to_bf16(acc3[r] * inv);
    }
}

// ---------------------------------------------------------------------------
extern "C" void kernel_launch(void* const* d_in, const int* in_sizes, int n_in,
                              void* d_out, int out_size, void* d_ws, size_t ws_size,
                              hipStream_t stream) {
    const float* x    = (const float*)d_in[0];
    const float* cosb = (const float*)d_in[1];
    const float* sinb = (const float*)d_in[2];
    const float* wq   = (const float*)d_in[3];
    const float* wk   = (const float*)d_in[4];
    const float* wv   = (const float*)d_in[5];
    const float* wo   = (const float*)d_in[6];
    const float* wfc  = (const float*)d_in[7];
    const float* wpr  = (const float*)d_in[8];
    const float* g1   = (const float*)d_in[9];
    const float* g2   = (const float*)d_in[10];
    float* out = (float*)d_out;

    char* ws = (char*)d_ws;
    size_t o = 0;
    auto take = [&](size_t bytes) { char* p = ws + o; o += (bytes + 255) & ~(size_t)255; return p; };

    unsigned short* wqT  = (unsigned short*)take((size_t)C_ * C_ * 2);
    unsigned short* wkT  = (unsigned short*)take((size_t)C_ * C_ * 2);
    unsigned short* wvT  = (unsigned short*)take((size_t)C_ * C_ * 2);
    unsigned short* woT  = (unsigned short*)take((size_t)C_ * C_ * 2);
    unsigned short* wfcT = (unsigned short*)take((size_t)FF_ * C_ * 2);
    unsigned short* wprT = (unsigned short*)take((size_t)C_ * FF_ * 2);
    unsigned short* h1   = (unsigned short*)take((size_t)BT_ * C_ * 2);
    float* qf            = (float*)take((size_t)BT_ * C_ * 4);
    float* kf            = (float*)take((size_t)BT_ * C_ * 4);
    float* vf            = (float*)take((size_t)BT_ * C_ * 4);
    unsigned short* qbb  = (unsigned short*)take((size_t)BT_ * C_ * 2);
    unsigned short* kbb  = (unsigned short*)take((size_t)BT_ * C_ * 2);
    unsigned short* vbb  = (unsigned short*)take((size_t)BT_ * C_ * 2);
    unsigned short* ybb  = (unsigned short*)take((size_t)BT_ * C_ * 2);
    float* x1            = (float*)take((size_t)BT_ * C_ * 4);
    unsigned short* h2   = (unsigned short*)take((size_t)BT_ * C_ * 2);
    unsigned short* ffb  = (unsigned short*)take((size_t)BT_ * FF_ * 2);
    (void)ws_size;

    // 1) weight prep: transpose + bf16
    transpose_bf16<<<(C_ * C_) / 256, 256, 0, stream>>>(wq, wqT, C_, C_);
    transpose_bf16<<<(C_ * C_) / 256, 256, 0, stream>>>(wk, wkT, C_, C_);
    transpose_bf16<<<(C_ * C_) / 256, 256, 0, stream>>>(wv, wvT, C_, C_);
    transpose_bf16<<<(C_ * C_) / 256, 256, 0, stream>>>(wo, woT, C_, C_);
    transpose_bf16<<<(C_ * FF_) / 256, 256, 0, stream>>>(wfc, wfcT, C_, FF_);  // Wt: FF x C
    transpose_bf16<<<(FF_ * C_) / 256, 256, 0, stream>>>(wpr, wprT, FF_, C_);  // Wt: C x FF

    // 2) h = rmsnorm(x, g1) -> bf16
    rmsnorm_bf16<<<BT_, 256, 0, stream>>>(x, g1, h1);

    // 3) q/k/v = h @ w{q,k,v}  (f32 out)
    dim3 gqkv(C_ / 64, BT_ / 128);
    wmma_gemm<<<gqkv, 256, 0, stream>>>(h1, wqT, nullptr, qf, BT_, C_, C_, 0);
    wmma_gemm<<<gqkv, 256, 0, stream>>>(h1, wkT, nullptr, kf, BT_, C_, C_, 0);
    wmma_gemm<<<gqkv, 256, 0, stream>>>(h1, wvT, nullptr, vf, BT_, C_, C_, 0);

    // 4) rope(q,k) -> bf16 (scale folded into q); v -> bf16
    rope_qk<<<(B_ * T_ * H_ * D2_) / 256, 256, 0, stream>>>(qf, kf, cosb, sinb, qbb, kbb);
    cvt_bf16<<<(BT_ * C_) / 256, 256, 0, stream>>>(vf, vbb, BT_ * C_);

    // 5) flash attention -> y bf16 (B,T,C)
    attn_kernel<<<dim3(B_ * H_, T_ / 128), 256, 0, stream>>>(qbb, kbb, vbb, ybb);

    // 6) x1 = x + y @ wo
    wmma_gemm<<<gqkv, 256, 0, stream>>>(ybb, woT, x, x1, BT_, C_, C_, 1);

    // 7) h2 = rmsnorm(x1, g2) -> bf16
    rmsnorm_bf16<<<BT_, 256, 0, stream>>>(x1, g2, h2);

    // 8) u = relu(h2 @ wfc)^2 -> bf16
    wmma_gemm<<<dim3(FF_ / 64, BT_ / 128), 256, 0, stream>>>(h2, wfcT, nullptr, ffb, BT_, FF_, C_, 2);

    // 9) out = x1 + u @ wpr
    wmma_gemm<<<gqkv, 256, 0, stream>>>(ffb, wprT, x1, out, BT_, C_, FF_, 1);
}
